// GroupedQueryAttention_25314537242695
// MI455X (gfx1250) — matrix operand order repaired and checked by reading, not compiled, not run on hardware
//
#include <hip/hip_runtime.h>
#include <hip/hip_bf16.h>
#include <math.h>

// ---------------------------------------------------------------------------
// GQA forward for MI455X (gfx1250, wave32, WMMA).
// All matmuls in f16 via v_wmma_f32_16x16x32_f16 (f32 accumulate).
// Flash-attention style online softmax, per-wave 16-row Q tiles.
// ---------------------------------------------------------------------------

typedef _Float16 h16;
typedef __attribute__((ext_vector_type(16))) _Float16 v16h;
typedef __attribute__((ext_vector_type(8)))  _Float16 v8h;
typedef __attribute__((ext_vector_type(8)))  float    v8f;

constexpr int SEQ = 2048;
constexpr int HID = 1024;
constexpr int NH  = 16;
constexpr int NKV = 4;
constexpr int HD  = 64;
constexpr float SCALING = 0.125f;
constexpr float NEGMIN  = -3.4028234663852886e38f;

// ---- workspace byte offsets (all 512KB-aligned) -----------------------------
constexpr size_t OFF_HF16 = 0;                        // 2048*1024 f16 = 4MB
constexpr size_t OFF_WQT  = 4u  << 20;                // 1024*1024 f16 = 2MB
constexpr size_t OFF_WKT  = 6u  << 20;                // 256*1024 f16 = .5MB
constexpr size_t OFF_WVT  = 6u  << 20 | (512u << 10); // .5MB
constexpr size_t OFF_WOT  = 7u  << 20;                // 2MB
constexpr size_t OFF_QRAW = 9u  << 20;                // 2048*1024 f32 = 8MB
constexpr size_t OFF_KRAW = 17u << 20;                // 2048*256 f32 = 2MB
constexpr size_t OFF_VRAW = 19u << 20;                // 2MB
constexpr size_t OFF_QF16 = 21u << 20;                // [NH][S][HD] f16 = 4MB
constexpr size_t OFF_KF16 = 25u << 20;                // [NKV][S][HD] f16 = 1MB
constexpr size_t OFF_VT   = 26u << 20;                // [NKV][HD][S] f16 = 1MB
constexpr size_t OFF_CTX  = 27u << 20;                // [S][NH*HD] f16 = 4MB

// ---------------------------------------------------------------------------
// WMMA fragment loaders (contiguous k order, same in A and B: lane half h holds k = 16h..16h+15)
// ---------------------------------------------------------------------------

// A 16x32 f16 from row-major [m][k] with stride lda.
// lane L: m = L%16, kh = L/16; halves 0..15 = k in [kh*16, kh*16+16).
__device__ __forceinline__ v16h load_a_frag(const h16* __restrict__ A, int lda) {
  const int lane = threadIdx.x & 31;
  const int m  = lane & 15;
  const int kh = lane >> 4;
  return *(const v16h*)(A + (size_t)m * lda + kh * 16);
}

// B 32x16 f16 from B^T stored row-major [n][k] with stride ldb.
// lane L: n = L%16, kh = L/16; halves 0..15 = k in [kh*16, kh*16+16).
__device__ __forceinline__ v16h load_b_frag(const h16* __restrict__ BT, int ldb) {
  const int lane = threadIdx.x & 31;
  const int n  = lane & 15;
  const int kh = lane >> 4;
  return *(const v16h*)(BT + (size_t)n * ldb + kh * 16);
}

#define WMMA_F16(a, b, c) \
  __builtin_amdgcn_wmma_f32_16x16x32_f16(false, (a), false, (b), (short)0, (c), false, false)

// ---------------------------------------------------------------------------
// Elementwise helpers
// ---------------------------------------------------------------------------
__global__ void k_f32_to_f16(const float* __restrict__ src, h16* __restrict__ dst, int n) {
  int i = blockIdx.x * blockDim.x + threadIdx.x;
  if (i < n) dst[i] = (h16)src[i];
}

// W [K][N] f32 -> Wt [N][K] f16
__global__ void k_transpose_f16(const float* __restrict__ src, h16* __restrict__ dst,
                                int K, int N) {
  int i = blockIdx.x * blockDim.x + threadIdx.x;
  if (i < N * K) {
    int n = i / K, k = i % K;
    dst[i] = (h16)src[(size_t)k * N + n];
  }
}

// ---------------------------------------------------------------------------
// Generic WMMA GEMM: C[M][N] f32 = A[M][K] f16 (row-major) x BT[N][K] f16
// block = 256 threads (8 waves); block tile 128(M) x 128(N); wave tile 32x64
// (2 m-frags x 4 n-frags = 8 WMMA per 32-deep k-step).
// ---------------------------------------------------------------------------
__global__ void k_gemm_f16f32(const h16* __restrict__ A, const h16* __restrict__ BT,
                              float* __restrict__ C, int M, int N, int K) {
  const int wave = threadIdx.x >> 5;
  const int lane = threadIdx.x & 31;
  const int wm = wave & 3;                 // 4 waves along M (4 * 32 = 128)
  const int wn = wave >> 2;                // 2 waves along N (2 * 64 = 128)
  const int m0 = blockIdx.y * 128 + wm * 32;
  const int n0 = blockIdx.x * 128 + wn * 64;

  v8f acc[2][4] = {};
  for (int k = 0; k < K; k += 32) {
    // lookahead prefetch (global_prefetch_b8): lane-strided over tile rows
    if (k + 256 < K) {
      __builtin_prefetch(A  + (size_t)(m0 + lane) * K + k + 256, 0, 3);
      __builtin_prefetch(BT + (size_t)(n0 + lane) * K + k + 256, 0, 3);
      __builtin_prefetch(BT + (size_t)(n0 + 32 + lane) * K + k + 256, 0, 3);
    }
    v16h a0 = load_a_frag(A + (size_t)m0 * K + k, K);
    v16h a1 = load_a_frag(A + (size_t)(m0 + 16) * K + k, K);
#pragma unroll
    for (int nj = 0; nj < 4; ++nj) {
      v16h b = load_b_frag(BT + (size_t)(n0 + nj * 16) * K + k, K);
      acc[0][nj] = WMMA_F16(a0, b, acc[0][nj]);
      acc[1][nj] = WMMA_F16(a1, b, acc[1][nj]);
    }
  }

  const int cn = lane & 15;
  const int mh = lane >> 4;
#pragma unroll
  for (int mi = 0; mi < 2; ++mi)
#pragma unroll
    for (int nj = 0; nj < 4; ++nj)
#pragma unroll
      for (int r = 0; r < 8; ++r) {
        int row = m0 + mi * 16 + r + 8 * mh;
        int col = n0 + nj * 16 + cn;
        C[(size_t)row * N + col] = acc[mi][nj][r];
      }
}

// ---------------------------------------------------------------------------
// RoPE + repack. Qraw[s][NH*HD], Kraw[s][NKV*HD], Vraw[s][NKV*HD] (f32) ->
//   Qf16 [h][s][d], Kf16 [kvh][s][d], Vt [kvh][d][s] (f16)
// ---------------------------------------------------------------------------
__global__ void k_rope_pack(const float* __restrict__ Qr, const float* __restrict__ Kr,
                            const float* __restrict__ Vr, const float* __restrict__ cosT,
                            const float* __restrict__ sinT, h16* __restrict__ Qo,
                            h16* __restrict__ Ko, h16* __restrict__ Vto) {
  const int C = (NH + 2 * NKV) * HD;  // 1536
  int i = blockIdx.x * blockDim.x + threadIdx.x;
  if (i >= SEQ * C) return;
  int s = i / C, c = i % C;
  if (c < NH * HD) {
    int hh = c / HD, d = c % HD;
    float x = Qr[(size_t)s * NH * HD + c];
    float xp = (d < HD / 2) ? -Qr[(size_t)s * NH * HD + hh * HD + d + HD / 2]
                            :  Qr[(size_t)s * NH * HD + hh * HD + d - HD / 2];
    float o = x * cosT[s * HD + d] + xp * sinT[s * HD + d];
    Qo[((size_t)hh * SEQ + s) * HD + d] = (h16)o;
  } else if (c < (NH + NKV) * HD) {
    int c2 = c - NH * HD;
    int hh = c2 / HD, d = c2 % HD;
    float x = Kr[(size_t)s * NKV * HD + c2];
    float xp = (d < HD / 2) ? -Kr[(size_t)s * NKV * HD + hh * HD + d + HD / 2]
                            :  Kr[(size_t)s * NKV * HD + hh * HD + d - HD / 2];
    float o = x * cosT[s * HD + d] + xp * sinT[s * HD + d];
    Ko[((size_t)hh * SEQ + s) * HD + d] = (h16)o;
  } else {
    int c3 = c - (NH + NKV) * HD;
    int hh = c3 / HD, d = c3 % HD;
    Vto[((size_t)(hh * HD + d)) * SEQ + s] = (h16)Vr[(size_t)s * NKV * HD + c3];
  }
}

// ---------------------------------------------------------------------------
// Flash attention: one wave = one (head, 16-row q tile); streams 32-key tiles.
// Q [h][s][d] f16, K [kvh][s][d] f16 (doubles as B^T for QK^T),
// V^T [kvh][d][s] f16 (B^T for P*V). Output ctx [s][NH*HD] f16.
// grid = (S/64, NH), block = 128 (4 waves).
// ---------------------------------------------------------------------------
__global__ void k_flash_attn(const h16* __restrict__ Q, const h16* __restrict__ Kt,
                             const h16* __restrict__ Vt,
                             const unsigned char* __restrict__ mask,
                             h16* __restrict__ ctx) {
  __shared__ __align__(64) h16 plds[4][16 * 32];  // per-wave P staging

  const int wave = threadIdx.x >> 5;
  const int lane = threadIdx.x & 31;
  const int cn = lane & 15;
  const int mh = lane >> 4;
  const int h   = blockIdx.y;
  const int kvh = h >> 2;  // NH/NKV = 4 consecutive q-heads per kv-head
  const int q0  = (blockIdx.x * 4 + wave) * 16;

  const h16* Qp = Q + ((size_t)h * SEQ + q0) * HD;
  v16h aq0 = load_a_frag(Qp, HD);        // k = 0..31 of head dim
  v16h aq1 = load_a_frag(Qp + 32, HD);   // k = 32..63

  v8f acc[4] = {};               // 16 x 64 output tile, 4 n-fragments
  float mrow[8], lrow[8];        // stats for rows r + 8*mh
#pragma unroll
  for (int r = 0; r < 8; ++r) { mrow[r] = -INFINITY; lrow[r] = 0.f; }

  for (int key0 = 0; key0 < SEQ; key0 += 32) {
    const h16* Kp = Kt + ((size_t)kvh * SEQ + key0) * HD;
    // prefetch next key tile (K rows + V rows) while this one computes
    if (key0 + 32 < SEQ) {
      __builtin_prefetch(Kp + (size_t)32 * HD + lane * HD, 0, 3);
      __builtin_prefetch(Vt + ((size_t)(kvh * HD + lane)) * SEQ + key0 + 32, 0, 3);
    }

    // ---- S = (Q K^T) * scale + mask, 16x32 as two 16x16 f32 fragments ----
    v16h b00 = load_b_frag(Kp, HD);
    v16h b01 = load_b_frag(Kp + 32, HD);
    v16h b10 = load_b_frag(Kp + 16 * HD, HD);
    v16h b11 = load_b_frag(Kp + 16 * HD + 32, HD);
    v8f s0 = {}, s1 = {};
    s0 = WMMA_F16(aq0, b00, s0);
    s0 = WMMA_F16(aq1, b01, s0);
    s1 = WMMA_F16(aq0, b10, s1);
    s1 = WMMA_F16(aq1, b11, s1);

    float pen0 = mask[key0 + cn]      ? 0.f : NEGMIN;
    float pen1 = mask[key0 + 16 + cn] ? 0.f : NEGMIN;

    // ---- online softmax: row reductions across the 16-lane half ----
    float p0[8], p1[8], alpha[8];
#pragma unroll
    for (int r = 0; r < 8; ++r) {
      float x0 = s0[r] * SCALING + pen0;
      float x1 = s1[r] * SCALING + pen1;
      float tm = fmaxf(x0, x1);
      tm = fmaxf(tm, __shfl_xor(tm, 1, 32));
      tm = fmaxf(tm, __shfl_xor(tm, 2, 32));
      tm = fmaxf(tm, __shfl_xor(tm, 4, 32));
      tm = fmaxf(tm, __shfl_xor(tm, 8, 32));
      float mnew = fmaxf(mrow[r], tm);
      float a = __expf(mrow[r] - mnew);   // first iter: exp(-inf)=0
      mrow[r] = mnew;
      float e0 = __expf(x0 - mnew);
      float e1 = __expf(x1 - mnew);
      float ts = e0 + e1;
      ts += __shfl_xor(ts, 1, 32);
      ts += __shfl_xor(ts, 2, 32);
      ts += __shfl_xor(ts, 4, 32);
      ts += __shfl_xor(ts, 8, 32);
      lrow[r] = lrow[r] * a + ts;
      alpha[r] = a; p0[r] = e0; p1[r] = e1;
    }

    // ---- rescale accumulator ----
#pragma unroll
    for (int nf = 0; nf < 4; ++nf)
#pragma unroll
      for (int r = 0; r < 8; ++r) acc[nf][r] *= alpha[r];

    // ---- stage P (D-layout) -> LDS row-major [16][32] -> A fragment ----
    h16* pw = plds[wave];
#pragma unroll
    for (int r = 0; r < 8; ++r) {
      int row = r + 8 * mh;
      pw[row * 32 + cn]      = (h16)p0[r];
      pw[row * 32 + 16 + cn] = (h16)p1[r];
    }
    asm volatile("s_wait_dscnt 0" ::: "memory");  // per-wave LDS RAW fence
    v16h ap = load_a_frag(pw, 32);

    // ---- acc += P (16x32) * V-tile (32x64) ----
#pragma unroll
    for (int nf = 0; nf < 4; ++nf) {
      const h16* Vp = Vt + ((size_t)(kvh * HD + nf * 16)) * SEQ + key0;
      v16h bv = load_b_frag(Vp, SEQ);
      acc[nf] = WMMA_F16(ap, bv, acc[nf]);
    }
  }

  // ---- epilogue: ctx[s][h*64+d] = acc / l ----
#pragma unroll
  for (int nf = 0; nf < 4; ++nf)
#pragma unroll
    for (int r = 0; r < 8; ++r) {
      int row = q0 + r + 8 * mh;
      float inv = 1.0f / lrow[r];
      ctx[(size_t)row * (NH * HD) + h * HD + nf * 16 + cn] = (h16)(acc[nf][r] * inv);
    }
}

// ---------------------------------------------------------------------------
extern "C" void kernel_launch(void* const* d_in, const int* in_sizes, int n_in,
                              void* d_out, int out_size, void* d_ws, size_t ws_size,
                              hipStream_t stream) {
  (void)in_sizes; (void)n_in; (void)out_size; (void)ws_size;
  const float* Hs   = (const float*)d_in[0];
  const float* cosT = (const float*)d_in[1];
  const float* sinT = (const float*)d_in[2];
  const unsigned char* msk = (const unsigned char*)d_in[3];  // bool (1B) mask
  const float* Wq = (const float*)d_in[4];
  const float* Wk = (const float*)d_in[5];
  const float* Wv = (const float*)d_in[6];
  const float* Wo = (const float*)d_in[7];
  float* out = (float*)d_out;

  char* ws = (char*)d_ws;
  h16*   Hf   = (h16*)(ws + OFF_HF16);
  h16*   Wqt  = (h16*)(ws + OFF_WQT);
  h16*   Wkt  = (h16*)(ws + OFF_WKT);
  h16*   Wvt  = (h16*)(ws + OFF_WVT);
  h16*   Wot  = (h16*)(ws + OFF_WOT);
  float* Qraw = (float*)(ws + OFF_QRAW);
  float* Kraw = (float*)(ws + OFF_KRAW);
  float* Vraw = (float*)(ws + OFF_VRAW);
  h16*   Qf   = (h16*)(ws + OFF_QF16);
  h16*   Kf   = (h16*)(ws + OFF_KF16);
  h16*   Vtt  = (h16*)(ws + OFF_VT);
  h16*   ctx  = (h16*)(ws + OFF_CTX);

  // 1) converts / weight transposes (f32 -> f16, W[K][N] -> Wt[N][K])
  k_f32_to_f16<<<(SEQ * HID + 255) / 256, 256, 0, stream>>>(Hs, Hf, SEQ * HID);
  k_transpose_f16<<<(HID * HID + 255) / 256, 256, 0, stream>>>(Wq, Wqt, HID, HID);
  k_transpose_f16<<<(HID * NKV * HD + 255) / 256, 256, 0, stream>>>(Wk, Wkt, HID, NKV * HD);
  k_transpose_f16<<<(HID * NKV * HD + 255) / 256, 256, 0, stream>>>(Wv, Wvt, HID, NKV * HD);
  k_transpose_f16<<<(HID * HID + 255) / 256, 256, 0, stream>>>(Wo, Wot, HID, HID);

  // 2) QKV projections (WMMA)
  k_gemm_f16f32<<<dim3(HID / 128, SEQ / 128), 256, 0, stream>>>(Hf, Wqt, Qraw, SEQ, HID, HID);
  k_gemm_f16f32<<<dim3(NKV * HD / 128, SEQ / 128), 256, 0, stream>>>(Hf, Wkt, Kraw, SEQ, NKV * HD, HID);
  k_gemm_f16f32<<<dim3(NKV * HD / 128, SEQ / 128), 256, 0, stream>>>(Hf, Wvt, Vraw, SEQ, NKV * HD, HID);

  // 3) RoPE + head-major repack (+ V transpose)
  k_rope_pack<<<(SEQ * (NH + 2 * NKV) * HD + 255) / 256, 256, 0, stream>>>(
      Qraw, Kraw, Vraw, cosT, sinT, Qf, Kf, Vtt);

  // 4) flash attention (WMMA)
  k_flash_attn<<<dim3(SEQ / 64, NH), 128, 0, stream>>>(Qf, Kf, Vtt, msk, ctx);

  // 5) output projection (WMMA) -> f32 d_out
  k_gemm_f16f32<<<dim3(HID / 128, SEQ / 128), 256, 0, stream>>>(ctx, Wot, out, SEQ, HID, HID);
}
